// HardQuadTripletRadiusSOSRLoss_40767829573847
// MI455X (gfx1250) — compile-verified, matrix operand order checked by reading, not run on hardware
//
#include <hip/hip_runtime.h>
#include <math.h>

typedef __attribute__((ext_vector_type(16))) _Float16 v16h;
typedef __attribute__((ext_vector_type(8)))  float    v8f;

#define GRIDSZ 8.0f
#define RAD2   256.0f      // (2*GRID_SIZE)^2
#define MARGIN_ 1.0f
#define BIGV   5.0f
#define HUGEV  1.0e6f
#define FH 120
#define FW 160
#define CCH 128

union F16Frag { v16h v; uint4 q[2]; };

// sorted ascending top-4 insert, compare+select (no NaN-canonicalize traffic)
__device__ __forceinline__ void insert4(float* t, float x) {
  bool c3 = x < t[3]; t[3] = c3 ? x : t[3];
  bool c2 = t[3] < t[2]; float lo = c2 ? t[3] : t[2]; float hi = c2 ? t[2] : t[3];
  t[2] = lo; t[3] = hi;
  bool c1 = t[2] < t[1]; lo = c1 ? t[2] : t[1]; hi = c1 ? t[1] : t[2];
  t[1] = lo; t[2] = hi;
  bool c0 = t[1] < t[0]; lo = c0 ? t[1] : t[0]; hi = c0 ? t[0] : t[1];
  t[0] = lo; t[1] = hi;
}

// sorted ascending key list with payload, branchless cascade insert
__device__ __forceinline__ void insert8p(float* k, float* p, float key, float val) {
#pragma unroll
  for (int j = 7; j >= 1; --j) {
    bool cj  = key < k[j];
    bool cjm = key < k[j-1];
    float nk = cjm ? k[j-1] : key;
    float np = cjm ? p[j-1] : val;
    k[j] = cj ? nk : k[j];
    p[j] = cj ? np : p[j];
  }
  bool c0 = key < k[0];
  k[0] = c0 ? key : k[0];
  p[0] = c0 ? val : p[0];
}

// ---------------- preprocess: normalize d1, bilinear+normalize w_desc, pos, vis ----
__global__ __launch_bounds__(256)
void prep_kernel(const float* __restrict__ kp1, const float* __restrict__ wkp1,
                 const float* __restrict__ kdesc, const float* __restrict__ desc2,
                 const float* __restrict__ homo,
                 _Float16* __restrict__ d1h, _Float16* __restrict__ wdh,
                 float* __restrict__ pos, float* __restrict__ vis,
                 float* __restrict__ accum, int N) {
  int lane = threadIdx.x & 31;
  int pt   = blockIdx.x * 8 + (threadIdx.x >> 5);
  int b    = pt / N;

  // normalize kp1_desc
  float dv[4];
  const float* kd = kdesc + (size_t)pt * CCH;
  float ss = 0.f;
#pragma unroll
  for (int j = 0; j < 4; ++j) { dv[j] = kd[lane + 32*j]; ss += dv[j]*dv[j]; }
#pragma unroll
  for (int off = 16; off >= 1; off >>= 1) ss += __shfl_xor(ss, off, 32);
  float inv1 = 1.f / (sqrtf(ss) + 1e-8f);
  _Float16* o1 = d1h + (size_t)pt * CCH;
#pragma unroll
  for (int j = 0; j < 4; ++j) o1[lane + 32*j] = (_Float16)(dv[j] * inv1);

  // bilinear sample desc2 at w_kp1
  float py = wkp1[(size_t)pt*2 + 0], px = wkp1[(size_t)pt*2 + 1];
  float gy = fminf(fmaxf(py / GRIDSZ - 0.5f, 0.f), (float)(FH - 1));
  float gx = fminf(fmaxf(px / GRIDSZ - 0.5f, 0.f), (float)(FW - 1));
  int y0 = min(max((int)floorf(gy), 0), FH - 2);
  int x0 = min(max((int)floorf(gx), 0), FW - 2);
  float wy = gy - (float)y0, wx = gx - (float)x0;
  const float* dp = desc2 + (size_t)b * CCH * (FH*FW);
  float sv[4]; float ss2 = 0.f;
#pragma unroll
  for (int j = 0; j < 4; ++j) {
    const float* ch = dp + (size_t)(lane + 32*j) * (FH*FW) + y0*FW + x0;
    float v00 = ch[0], v01 = ch[1], v10 = ch[FW], v11 = ch[FW+1];
    sv[j] = v00*(1.f-wy)*(1.f-wx) + v01*(1.f-wy)*wx + v10*wy*(1.f-wx) + v11*wy*wx;
    ss2 += sv[j]*sv[j];
  }
#pragma unroll
  for (int off = 16; off >= 1; off >>= 1) ss2 += __shfl_xor(ss2, off, 32);
  float inv2 = 1.f / (sqrtf(ss2) + 1e-8f);
  _Float16* o2 = wdh + (size_t)pt * CCH;
  float ds = 0.f;
#pragma unroll
  for (int j = 0; j < 4; ++j) {
    float a = dv[j]*inv1, c = sv[j]*inv2;
    o2[lane + 32*j] = (_Float16)c;
    float d = a - c; ds += d*d;
  }
#pragma unroll
  for (int off = 16; off >= 1; off >>= 1) ds += __shfl_xor(ds, off, 32);

  if (lane == 0) {
    pos[pt] = sqrtf(ds + 1e-12f);
    const float* Hm = homo + b*9;
    float ky = kp1[(size_t)pt*2+0], kx = kp1[(size_t)pt*2+1];   // stored (y,x)
    float q0 = Hm[0]*kx + Hm[1]*ky + Hm[2];
    float q1 = Hm[3]*kx + Hm[4]*ky + Hm[5];
    float q2 = Hm[6]*kx + Hm[7]*ky + Hm[8];
    float qx = q0 / (q2 + 1e-8f), qy = q1 / (q2 + 1e-8f);
    float vv = (qx >= 0.f && qx < 1280.f && qy >= 0.f && qy < 960.f) ? 1.f : 0.f;
    vis[pt] = vv;
    atomicAdd(&accum[0], vv);
  }
}

// ---- FOS helpers: prefetch one column tile's B-fragments + coords ----------------
__device__ __forceinline__ void load_btile(const _Float16* __restrict__ cd,
                                           const float* __restrict__ wc,
                                           int ct, int lane, int khi,
                                           F16Frag* Bf, float* cY, float* cX) {
  int col = (ct << 4) + (lane & 15);
  const _Float16* cb = cd + (size_t)col * CCH + khi;
#pragma unroll
  for (int t = 0; t < 4; ++t) {
    Bf[t].q[0] = *(const uint4*)(cb + 32*t);
    Bf[t].q[1] = *(const uint4*)(cb + 32*t + 16);
  }
  *cY = wc[(size_t)col*2 + 0];
  *cX = wc[(size_t)col*2 + 1];
}

__device__ __forceinline__ void fos_tile(const _Float16* ldsA, int lane, int khi,
                                         const F16Frag* Bf, float cY, float cX,
                                         const float* rY, const float* rX,
                                         float tk[8][4]) {
  v8f acc = {0,0,0,0,0,0,0,0};
#pragma unroll
  for (int t = 0; t < 4; ++t) {
    F16Frag A;
    int ao = (lane & 15) * CCH + 32*t + khi;
    A.q[0] = *(const uint4*)(ldsA + ao);
    A.q[1] = *(const uint4*)(ldsA + ao + 16);
    acc = __builtin_amdgcn_wmma_f32_16x16x32_f16(false, A.v, false, Bf[t].v,
                                                 (short)0, acc, false, false);
  }
#pragma unroll
  for (int v = 0; v < 8; ++v) {
    float d  = __builtin_amdgcn_sqrtf(fmaxf(2.f - 2.f*acc[v], 1e-12f));
    float dy = rY[v]-cY, dx = rX[v]-cX;
    float x  = (dy*dy + dx*dx <= RAD2) ? BIGV : d;
    insert4(tk[v], x);
  }
}

// ---------------- FOS: fused WMMA Gram + radius mask + per-row top-4 ---------------
// Software-pipelined: ping-pong B-fragment buffers so tile j+1 loads overlap tile j
// WMMA + selection epilogue.
__global__ __launch_bounds__(32)
void fos_kernel(const _Float16* __restrict__ rowDesc, const _Float16* __restrict__ colDesc,
                const float* __restrict__ wco, const float* __restrict__ pos,
                float* __restrict__ tout, int N) {
  int lane = threadIdx.x;
  int ntiles = N >> 4;
  int b = blockIdx.x / ntiles;
  int rbase = (blockIdx.x % ntiles) << 4;
  size_t dbase = (size_t)b * N * CCH;
  size_t cbase = (size_t)b * N * 2;
  const _Float16* cd = colDesc + dbase;
  const float*    wc = wco + cbase;

  __shared__ __align__(16) _Float16 ldsA[16 * CCH];
  {
    const uint4* src = (const uint4*)(rowDesc + dbase + (size_t)rbase * CCH);
    uint4* dst = (uint4*)ldsA;
#pragma unroll
    for (int i = 0; i < 8; ++i) dst[lane + 32*i] = src[lane + 32*i];
  }
  __syncthreads();

  float rY[8], rX[8];
  {
    float myY = wc[(size_t)(rbase + (lane & 15))*2 + 0];
    float myX = wc[(size_t)(rbase + (lane & 15))*2 + 1];
    int g = (lane >> 4) << 3;
#pragma unroll
    for (int v = 0; v < 8; ++v) { rY[v] = __shfl(myY, v+g, 32); rX[v] = __shfl(myX, v+g, 32); }
  }

  float tk[8][4];
#pragma unroll
  for (int v = 0; v < 8; ++v) { tk[v][0]=tk[v][1]=tk[v][2]=tk[v][3]=3.0e38f; }

  int khi = (lane < 16) ? 0 : 8;

  F16Frag B0[4], B1[4];
  float cY0, cX0, cY1, cX1;
  load_btile(cd, wc, 0, lane, khi, B0, &cY0, &cX0);
  load_btile(cd, wc, 1, lane, khi, B1, &cY1, &cX1);

  for (int ct = 0; ct < ntiles; ct += 2) {
    fos_tile(ldsA, lane, khi, B0, cY0, cX0, rY, rX, tk);
    int n0 = ct + 2; if (n0 >= ntiles) n0 = 0;      // tail prefetch wraps (discarded)
    load_btile(cd, wc, n0, lane, khi, B0, &cY0, &cX0);

    fos_tile(ldsA, lane, khi, B1, cY1, cX1, rY, rX, tk);
    int n1 = ct + 3; if (n1 >= ntiles) n1 = 0;
    load_btile(cd, wc, n1, lane, khi, B1, &cY1, &cX1);
  }

  // merge per-lane lists across the 16 lanes sharing each row
#pragma unroll
  for (int off = 1; off < 16; off <<= 1) {
#pragma unroll
    for (int v = 0; v < 8; ++v) {
      float o0 = __shfl_xor(tk[v][0], off, 32);
      float o1 = __shfl_xor(tk[v][1], off, 32);
      float o2 = __shfl_xor(tk[v][2], off, 32);
      float o3 = __shfl_xor(tk[v][3], off, 32);
      insert4(tk[v], o0); insert4(tk[v], o1); insert4(tk[v], o2); insert4(tk[v], o3);
    }
  }
  if ((lane & 15) == 0) {
    int ro = (lane >> 4) << 3;
#pragma unroll
    for (int v = 0; v < 8; ++v) {
      int M = rbase + ro + v;
      float pv = pos[b*N + M];
      float s = 0.f;
#pragma unroll
      for (int kk = 0; kk < 4; ++kk) s += fmaxf(MARGIN_ + pv - tk[v][kk], 0.f);
      tout[b*N + M] = 0.25f * s;
    }
  }
}

// ---------------- SOS: two WMMA Grams + validity mask + paired top-8 ---------------
__global__ __launch_bounds__(32)
void sos_kernel(const _Float16* __restrict__ d1h, const _Float16* __restrict__ wdh,
                const float* __restrict__ kco, const float* __restrict__ wco,
                float* __restrict__ sout, int N) {
  int lane = threadIdx.x;
  int ntiles = N >> 4;
  int b = blockIdx.x / ntiles;
  int rbase = (blockIdx.x % ntiles) << 4;
  size_t dbase = (size_t)b * N * CCH;
  size_t cbase = (size_t)b * N * 2;

  __shared__ __align__(16) _Float16 ldsA[16*CCH];
  __shared__ __align__(16) _Float16 ldsW[16*CCH];
  {
    const uint4* s1 = (const uint4*)(d1h + dbase + (size_t)rbase*CCH);
    const uint4* s2 = (const uint4*)(wdh + dbase + (size_t)rbase*CCH);
    uint4* t1p = (uint4*)ldsA; uint4* t2p = (uint4*)ldsW;
#pragma unroll
    for (int i = 0; i < 8; ++i) { t1p[lane+32*i]=s1[lane+32*i]; t2p[lane+32*i]=s2[lane+32*i]; }
  }
  __syncthreads();

  float rKY[8],rKX[8],rWY[8],rWX[8];
  {
    int rr = rbase + (lane & 15);
    float a = kco[cbase + (size_t)rr*2+0], c = kco[cbase + (size_t)rr*2+1];
    float e = wco[cbase + (size_t)rr*2+0], f = wco[cbase + (size_t)rr*2+1];
    int g = (lane >> 4) << 3;
#pragma unroll
    for (int v = 0; v < 8; ++v) {
      rKY[v]=__shfl(a,v+g,32); rKX[v]=__shfl(c,v+g,32);
      rWY[v]=__shfl(e,v+g,32); rWX[v]=__shfl(f,v+g,32);
    }
  }

  float key8[8][8], val8[8][8];
#pragma unroll
  for (int v = 0; v < 8; ++v)
#pragma unroll
    for (int kk = 0; kk < 8; ++kk) { key8[v][kk] = 3.0e38f; val8[v][kk] = 0.f; }

  int khi = (lane < 16) ? 0 : 8;
  for (int ct = 0; ct < ntiles; ++ct) {
    int col = (ct << 4) + (lane & 15);
    float cKY = kco[cbase + (size_t)col*2+0], cKX = kco[cbase + (size_t)col*2+1];
    float cWY = wco[cbase + (size_t)col*2+0], cWX = wco[cbase + (size_t)col*2+1];
    v8f a11 = {0,0,0,0,0,0,0,0};
    v8f a22 = {0,0,0,0,0,0,0,0};
#pragma unroll
    for (int t = 0; t < 4; ++t) {
      F16Frag A, Bf;
      int ao = (lane & 15) * CCH + 32*t + khi;
      size_t co = dbase + (size_t)col*CCH + 32*t + khi;
      A.q[0]  = *(const uint4*)(ldsA + ao); A.q[1]  = *(const uint4*)(ldsA + ao + 16);
      Bf.q[0] = *(const uint4*)(d1h + co); Bf.q[1] = *(const uint4*)(d1h + co + 16);
      a11 = __builtin_amdgcn_wmma_f32_16x16x32_f16(false, A.v, false, Bf.v, (short)0, a11, false, false);
      A.q[0]  = *(const uint4*)(ldsW + ao); A.q[1]  = *(const uint4*)(ldsW + ao + 16);
      Bf.q[0] = *(const uint4*)(wdh + co); Bf.q[1] = *(const uint4*)(wdh + co + 16);
      a22 = __builtin_amdgcn_wmma_f32_16x16x32_f16(false, A.v, false, Bf.v, (short)0, a22, false, false);
    }
#pragma unroll
    for (int v = 0; v < 8; ++v) {
      float d11 = __builtin_amdgcn_sqrtf(fmaxf(2.f - 2.f*a11[v], 1e-12f));
      float d22 = __builtin_amdgcn_sqrtf(fmaxf(2.f - 2.f*a22[v], 1e-12f));
      float dy1 = rKY[v]-cKY, dx1 = rKX[v]-cKX;
      float dy2 = rWY[v]-cWY, dx2 = rWX[v]-cWX;
      bool valid = (dy1*dy1+dx1*dx1 > RAD2) && (dy2*dy2+dx2*dx2 > RAD2);
      float score = valid ? d11 : HUGEV;
      insert8p(key8[v], val8[v], score, d22);
    }
  }
#pragma unroll
  for (int off = 1; off < 16; off <<= 1) {
#pragma unroll
    for (int v = 0; v < 8; ++v) {
      float ok[8], ov[8];
#pragma unroll
      for (int kk = 0; kk < 8; ++kk) {
        ok[kk] = __shfl_xor(key8[v][kk], off, 32);
        ov[kk] = __shfl_xor(val8[v][kk], off, 32);
      }
#pragma unroll
      for (int kk = 0; kk < 8; ++kk) insert8p(key8[v], val8[v], ok[kk], ov[kk]);
    }
  }
  if ((lane & 15) == 0) {
    int ro = (lane >> 4) << 3;
#pragma unroll
    for (int v = 0; v < 8; ++v) {
      float s = 0.f;
#pragma unroll
      for (int kk = 0; kk < 8; ++kk) {
        float sd = key8[v][kk];
        float df = sd - val8[v][kk];
        s += (sd < HUGEV * 0.1f) ? df*df : 0.f;
      }
      sout[b*N + rbase + ro + v] = sqrtf(s + 1e-12f);
    }
  }
}

// ---------------- reductions ----------------
__global__ void init_kernel(float* accum) { accum[0] = 0.f; accum[1] = 0.f; }

__global__ __launch_bounds__(256)
void reduce_kernel(const float* __restrict__ vis, const float* __restrict__ t1,
                   const float* __restrict__ t2, const float* __restrict__ sosv,
                   float* __restrict__ accum, int total) {
  __shared__ float sm[256];
  int tid = threadIdx.x;
  float s = 0.f;
  for (int i = blockIdx.x*256 + tid; i < total; i += gridDim.x*256)
    s += vis[i] * (0.5f*(t1[i]+t2[i]) + sosv[i]);
  sm[tid] = s; __syncthreads();
  for (int off = 128; off >= 1; off >>= 1) {
    if (tid < off) sm[tid] += sm[tid+off];
    __syncthreads();
  }
  if (tid == 0) atomicAdd(&accum[1], sm[0]);
}

__global__ void fin_kernel(const float* accum, float* out) {
  out[0] = accum[1] / (accum[0] + 1e-8f);   // LOSS_LAMBDA == 1
}

extern "C" void kernel_launch(void* const* d_in, const int* in_sizes, int n_in,
                              void* d_out, int out_size, void* d_ws, size_t ws_size,
                              hipStream_t stream) {
  const float* kp1   = (const float*)d_in[0];
  const float* wkp1  = (const float*)d_in[1];
  const float* kdesc = (const float*)d_in[2];
  const float* desc2 = (const float*)d_in[3];
  const float* homo  = (const float*)d_in[4];
  int B  = in_sizes[4] / 9;
  int BN = in_sizes[0] / 2;
  int N  = BN / B;

  char* ws = (char*)d_ws;
  size_t o = 0;
  _Float16* d1h = (_Float16*)(ws + o); o += (size_t)BN*CCH*sizeof(_Float16);
  _Float16* wdh = (_Float16*)(ws + o); o += (size_t)BN*CCH*sizeof(_Float16);
  float* pos  = (float*)(ws + o); o += (size_t)BN*4;
  float* vis  = (float*)(ws + o); o += (size_t)BN*4;
  float* t1   = (float*)(ws + o); o += (size_t)BN*4;
  float* t2   = (float*)(ws + o); o += (size_t)BN*4;
  float* sosv = (float*)(ws + o); o += (size_t)BN*4;
  float* accum= (float*)(ws + o); o += 256;

  hipLaunchKernelGGL(init_kernel, dim3(1), dim3(1), 0, stream, accum);
  hipLaunchKernelGGL(prep_kernel, dim3(BN/8), dim3(256), 0, stream,
                     kp1, wkp1, kdesc, desc2, homo, d1h, wdh, pos, vis, accum, N);
  int nt = B * (N >> 4);
  hipLaunchKernelGGL(fos_kernel, dim3(nt), dim3(32), 0, stream, d1h, wdh, wkp1, pos, t1, N);
  hipLaunchKernelGGL(fos_kernel, dim3(nt), dim3(32), 0, stream, wdh, d1h, wkp1, pos, t2, N);
  hipLaunchKernelGGL(sos_kernel, dim3(nt), dim3(32), 0, stream, d1h, wdh, kp1, wkp1, sosv, N);
  hipLaunchKernelGGL(reduce_kernel, dim3(32), dim3(256), 0, stream, vis, t1, t2, sosv, accum, BN);
  hipLaunchKernelGGL(fin_kernel, dim3(1), dim3(1), 0, stream, accum, (float*)d_out);
}
